// FoldedQuantizedBottleneck_30623116821393
// MI455X (gfx1250) — 8-bit matrix operand order repaired and checked by reading, compiled, not run on hardware
//
#include <hip/hip_runtime.h>

typedef __attribute__((ext_vector_type(8))) int v8i;

#define Z_IN_C   128
#define Z_TGT_C  8
#define M0_RS_C  1073741824ll
#define SH_RS_C  4
#define M0_C1_C  1288490189ll
#define SH_C1_C  12
#define Z3_C1_C  128
#define M0_C2_C  1288490189ll
#define SH_C2_C  10
#define Z3_C2_C  128
#define M0_C3_C  1288490189ll
#define SH_C3_C  10
#define Z3_C3_C  128
#define M0_B_C   1073741824ll
#define SH_B_C   1
#define M0_P_C   1073741824ll
#define SH_P_C   1
#define Z3_ADD_C 128

#define NPIX   50176
#define HWSZ   784
#define MTILES 3136

__device__ __forceinline__ long long mulM(long long x, long long m0) {
    long long prod  = x * m0;
    long long nudge = (prod >= 0) ? (1ll << 30) : (1ll - (1ll << 30));
    return (prod + nudge) >> 31;
}
__device__ __forceinline__ long long rshr(long long x, int s) {
    long long mask = (1ll << s) - 1;
    long long rem  = x & mask;
    long long thr  = (mask >> 1) + ((x < 0) ? 1ll : 0ll);
    return (x >> s) + ((rem > thr) ? 1ll : 0ll);
}
__device__ __forceinline__ int requant_u8(int acc, int bias, long long m0, int sh, int zout) {
    long long t = rshr(mulM((long long)acc + (long long)bias, m0), sh) + zout;
    t = (t < 0) ? 0 : (t > 255 ? 255 : t);
    return (int)t;
}

__global__ void pack_w_kernel(const float* __restrict__ w, unsigned char* __restrict__ Bp,
                              int ntiles, int ktiles, int mode) {
    int idx   = blockIdx.x * blockDim.x + threadIdx.x;
    int total = ntiles * ktiles * 32 * 8;
    if (idx >= total) return;
    int v     = idx & 7;
    int rest  = idx >> 3;
    int lane  = rest & 31;
    int rest2 = rest >> 5;
    int kt    = rest2 % ktiles;
    int nt    = rest2 / ktiles;
    int n     = nt * 16 + (lane & 15);
    int koff  = ((lane >> 4) << 3) + (v & 1) * 4 + ((v >> 1) << 4);
    int kbase = kt * 64 + koff;
    int K     = ktiles * 64;
    unsigned int word = 0;
#pragma unroll
    for (int j = 0; j < 4; ++j) {
        int k = kbase + j;
        int widx = (mode == 0) ? (n * K + k) : ((n * 128 + (k & 127)) * 9 + (k >> 7));
        int wq = (int)w[widx];
        word |= ((unsigned int)(wq & 0xff)) << (8 * j);
    }
    *(unsigned int*)(Bp + (size_t)idx * 4) = word;
}

__global__ void bias_prep_kernel(const float* __restrict__ b, const float* __restrict__ w,
                                 int* __restrict__ out, int N, int K, int zin) {
    int n = blockIdx.x * blockDim.x + threadIdx.x;
    if (n >= N) return;
    long long s = 0;
    for (int k = 0; k < K; ++k) s += (long long)(int)w[(size_t)n * K + k];
    out[n] = (int)((long long)(int)b[n] - (long long)zin * s);
}

__device__ __forceinline__ v8i load_a_frag(const unsigned char* arow, int kc) {
    union { v8i v; uint2 q[4]; } au;
#pragma unroll
    for (int i = 0; i < 4; ++i)
        au.q[i] = *(const uint2*)(arow + kc + i * 16);
    return au.v;
}
__device__ __forceinline__ v8i load_b_frag(const unsigned char* Bp, int tileIdx, int lane) {
    const uint4* bp = (const uint4*)(Bp + ((size_t)tileIdx * 32 + lane) * 32);
    union { v8i v; uint4 q[2]; } bu;
    bu.q[0] = bp[0];
    bu.q[1] = bp[1];
    return bu.v;
}

__global__ __launch_bounds__(128) void conv1_kernel(const int* __restrict__ x,
                                                    const unsigned char* __restrict__ Bp,
                                                    const int* __restrict__ bias,
                                                    unsigned char* __restrict__ act2,
                                                    signed char* __restrict__ x1q) {
    __shared__ __align__(16) unsigned char As[16 * 512];
    const int m0    = blockIdx.x * 16;
    const int n_img = m0 / HWSZ;
    const int hw    = m0 % HWSZ;

    {
        const int t  = threadIdx.x;
        const int m  = t & 15;
        const int k0 = t >> 4;
        for (int k = k0; k < 512; k += 8) {
            size_t idx = ((size_t)(n_img * 512 + k)) * HWSZ + hw + m;
            int xv = x[idx];
            long long d = (long long)(xv - Z_IN_C);
            long long q = rshr(mulM(d, M0_RS_C), SH_RS_C) + Z_TGT_C;
            q = (q < 0) ? 0 : (q > 15 ? 15 : q);
            As[m * 512 + k] = (unsigned char)q;
            x1q[idx] = (signed char)rshr(mulM(d, M0_B_C), SH_B_C);
        }
    }
    __syncthreads();

    const int lane = threadIdx.x & 31;
    const int wv   = threadIdx.x >> 5;
    const int nt0  = wv * 2, nt1 = nt0 + 1;
    const int mm   = lane & 15;
    const int kb   = (lane >> 4) << 3;
    const unsigned char* arow = As + mm * 512 + kb;

    v8i acc0 = {0, 0, 0, 0, 0, 0, 0, 0};
    v8i acc1 = {0, 0, 0, 0, 0, 0, 0, 0};
#pragma unroll
    for (int kc = 0; kc < 512; kc += 64) {
        v8i a  = load_a_frag(arow, kc);
        v8i b0 = load_b_frag(Bp, nt0 * 8 + (kc >> 6), lane);
        v8i b1 = load_b_frag(Bp, nt1 * 8 + (kc >> 6), lane);
        acc0 = __builtin_amdgcn_wmma_i32_16x16x64_iu8(false, a, true, b0, acc0, false, false);
        acc1 = __builtin_amdgcn_wmma_i32_16x16x64_iu8(false, a, true, b1, acc1, false, false);
    }

    const int c0 = nt0 * 16 + mm, c1 = nt1 * 16 + mm;
    const int mbase = (lane >> 4) * 8;
    const int bv0 = bias[c0], bv1 = bias[c1];
#pragma unroll
    for (int v = 0; v < 8; ++v) {
        size_t row = (size_t)(m0 + mbase + v) * 128;
        act2[row + c0] = (unsigned char)requant_u8(acc0[v], bv0, M0_C1_C, SH_C1_C, Z3_C1_C);
        act2[row + c1] = (unsigned char)requant_u8(acc1[v], bv1, M0_C2_C == 0 ? 0 : M0_C1_C, SH_C1_C, Z3_C1_C);
    }
}

__global__ __launch_bounds__(128) void conv2_kernel(const unsigned char* __restrict__ act2,
                                                    const unsigned char* __restrict__ Bp,
                                                    const int* __restrict__ bias,
                                                    unsigned char* __restrict__ act3) {
    __shared__ __align__(16) unsigned char As[16 * 1152];
    const int m0    = blockIdx.x * 16;
    const int n_img = m0 / HWSZ;
    const int hw    = m0 % HWSZ;

    {
        const int t   = threadIdx.x;
        const int m   = t & 15;
        const int hwm = hw + m;
        const int h   = hwm / 28;
        const int w   = hwm % 28;
        for (int d = (t >> 4); d < 288; d += 8) {
            int tap = d >> 5;
            int ch4 = (d & 31) * 4;
            int dy  = tap / 3 - 1;
            int dx  = tap % 3 - 1;
            int hh  = h + dy;
            int ww  = w + dx;
            unsigned int val = 0x80808080u;
            if ((unsigned)hh < 28u && (unsigned)ww < 28u)
                val = *(const unsigned int*)(act2 +
                        ((size_t)(n_img * HWSZ + hh * 28 + ww)) * 128 + ch4);
            *(unsigned int*)(As + m * 1152 + d * 4) = val;
        }
    }
    __syncthreads();

    const int lane = threadIdx.x & 31;
    const int wv   = threadIdx.x >> 5;
    const int nt0  = wv * 2, nt1 = nt0 + 1;
    const int mm   = lane & 15;
    const int kb   = (lane >> 4) << 3;
    const unsigned char* arow = As + mm * 1152 + kb;

    v8i acc0 = {0, 0, 0, 0, 0, 0, 0, 0};
    v8i acc1 = {0, 0, 0, 0, 0, 0, 0, 0};
#pragma unroll 3
    for (int kt = 0; kt < 18; ++kt) {
        v8i a  = load_a_frag(arow, kt * 64);
        v8i b0 = load_b_frag(Bp, nt0 * 18 + kt, lane);
        v8i b1 = load_b_frag(Bp, nt1 * 18 + kt, lane);
        acc0 = __builtin_amdgcn_wmma_i32_16x16x64_iu8(false, a, true, b0, acc0, false, false);
        acc1 = __builtin_amdgcn_wmma_i32_16x16x64_iu8(false, a, true, b1, acc1, false, false);
    }

    const int c0 = nt0 * 16 + mm, c1 = nt1 * 16 + mm;
    const int mbase = (lane >> 4) * 8;
    const int bv0 = bias[c0], bv1 = bias[c1];
#pragma unroll
    for (int v = 0; v < 8; ++v) {
        size_t row = (size_t)(m0 + mbase + v) * 128;
        act3[row + c0] = (unsigned char)requant_u8(acc0[v], bv0, M0_C2_C, SH_C2_C, Z3_C2_C);
        act3[row + c1] = (unsigned char)requant_u8(acc1[v], bv1, M0_C2_C, SH_C2_C, Z3_C2_C);
    }
}

__global__ __launch_bounds__(128) void conv3_kernel(const unsigned char* __restrict__ act3,
                                                    const unsigned char* __restrict__ Bp,
                                                    const int* __restrict__ bias,
                                                    const signed char* __restrict__ x1q,
                                                    int* __restrict__ out) {
    __shared__ __align__(16) unsigned char As[16 * 128];
    __shared__ int tbuf[4][2][16][16];
    const int m0    = blockIdx.x * 16;
    const int n_img = m0 / HWSZ;
    const int hw    = m0 % HWSZ;

    ((uint4*)As)[threadIdx.x] = ((const uint4*)(act3 + (size_t)m0 * 128))[threadIdx.x];
    __syncthreads();

    const int lane = threadIdx.x & 31;
    const int wv   = threadIdx.x >> 5;
    const int mm   = lane & 15;
    const int kb   = (lane >> 4) << 3;
    const unsigned char* arow = As + mm * 128 + kb;

    const v8i a0 = load_a_frag(arow, 0);
    const v8i a1 = load_a_frag(arow, 64);
    const int mbase = (lane >> 4) * 8;
    const int tt = lane >> 4;
    const int cl = lane & 15;

    for (int j = 0; j < 4; ++j) {
        const int nt0 = wv * 8 + j * 2, nt1 = nt0 + 1;
        v8i acc0 = {0, 0, 0, 0, 0, 0, 0, 0};
        v8i acc1 = {0, 0, 0, 0, 0, 0, 0, 0};
        v8i b00 = load_b_frag(Bp, nt0 * 2 + 0, lane);
        v8i b10 = load_b_frag(Bp, nt1 * 2 + 0, lane);
        v8i b01 = load_b_frag(Bp, nt0 * 2 + 1, lane);
        v8i b11 = load_b_frag(Bp, nt1 * 2 + 1, lane);
        acc0 = __builtin_amdgcn_wmma_i32_16x16x64_iu8(false, a0, true, b00, acc0, false, false);
        acc1 = __builtin_amdgcn_wmma_i32_16x16x64_iu8(false, a0, true, b10, acc1, false, false);
        acc0 = __builtin_amdgcn_wmma_i32_16x16x64_iu8(false, a1, true, b01, acc0, false, false);
        acc1 = __builtin_amdgcn_wmma_i32_16x16x64_iu8(false, a1, true, b11, acc1, false, false);

        const int bv0 = bias[nt0 * 16 + mm], bv1 = bias[nt1 * 16 + mm];
#pragma unroll
        for (int v = 0; v < 8; ++v) {
            tbuf[wv][0][mbase + v][mm] = requant_u8(acc0[v], bv0, M0_C3_C, SH_C3_C, Z3_C3_C);
            tbuf[wv][1][mbase + v][mm] = requant_u8(acc1[v], bv1, M0_C3_C, SH_C3_C, Z3_C3_C);
        }
        __syncthreads();

        const int c = (nt0 + tt) * 16 + cl;
        const size_t base = ((size_t)(n_img * 512 + c)) * HWSZ + hw;
        union { uint4 u; signed char b[16]; } xw;
        xw.u = *(const uint4*)(x1q + base);
        int res[16];
#pragma unroll
        for (int m = 0; m < 16; ++m) {
            long long x2 = rshr(mulM((long long)tbuf[wv][tt][m][cl] - Z3_C3_C, M0_P_C), SH_P_C);
            long long o  = (long long)xw.b[m] + x2 + Z3_ADD_C;
            res[m] = (int)((o < 0) ? 0 : (o > 255 ? 255 : o));
        }
#pragma unroll
        for (int q = 0; q < 4; ++q) {
            int4 r;
            r.x = res[q * 4 + 0]; r.y = res[q * 4 + 1];
            r.z = res[q * 4 + 2]; r.w = res[q * 4 + 3];
            ((int4*)(out + base))[q] = r;
        }
        __syncthreads();
    }
}

extern "C" void kernel_launch(void* const* d_in, const int* in_sizes, int n_in,
                              void* d_out, int out_size, void* d_ws, size_t ws_size,
                              hipStream_t stream) {
    (void)in_sizes; (void)n_in; (void)out_size; (void)ws_size;
    const int*   x  = (const int*)d_in[0];
    const float* w1 = (const float*)d_in[1];
    const float* b1 = (const float*)d_in[2];
    const float* w2 = (const float*)d_in[3];
    const float* b2 = (const float*)d_in[4];
    const float* w3 = (const float*)d_in[5];
    const float* b3 = (const float*)d_in[6];
    int* out = (int*)d_out;

    unsigned char* ws = (unsigned char*)d_ws;
    const size_t ACT  = (size_t)NPIX * 128;
    const size_t XQ   = (size_t)NPIX * 512;
    unsigned char* act2 = ws;
    unsigned char* act3 = ws + ACT;
    signed char*   x1q  = (signed char*)(ws + 2 * ACT);
    unsigned char* Bp1  = ws + 2 * ACT + XQ;
    unsigned char* Bp2  = Bp1 + 8 * 8 * 1024;
    unsigned char* Bp3  = Bp2 + 8 * 18 * 1024;
    int* bias1 = (int*)(Bp3 + 32 * 2 * 1024);
    int* bias2 = bias1 + 128;
    int* bias3 = bias2 + 128;

    pack_w_kernel<<<(8 * 8 * 256 + 255) / 256, 256, 0, stream>>>(w1, Bp1, 8, 8, 0);
    pack_w_kernel<<<(8 * 18 * 256 + 255) / 256, 256, 0, stream>>>(w2, Bp2, 8, 18, 1);
    pack_w_kernel<<<(32 * 2 * 256 + 255) / 256, 256, 0, stream>>>(w3, Bp3, 32, 2, 0);
    bias_prep_kernel<<<1, 128, 0, stream>>>(b1, w1, bias1, 128, 512, Z_TGT_C);
    bias_prep_kernel<<<1, 128, 0, stream>>>(b2, w2, bias2, 128, 1152, Z3_C1_C);
    bias_prep_kernel<<<2, 256, 0, stream>>>(b3, w3, bias3, 512, 128, Z3_C2_C);

    conv1_kernel<<<MTILES, 128, 0, stream>>>(x, Bp1, bias1, act2, x1q);
    conv2_kernel<<<MTILES, 128, 0, stream>>>(act2, Bp2, bias2, act3);
    conv3_kernel<<<MTILES, 128, 0, stream>>>(act3, Bp3, bias3, x1q, out);
}
